// decoder_40733469835931
// MI455X (gfx1250) — matrix operand order repaired and checked by reading, not compiled, not run on hardware
//
#include <hip/hip_runtime.h>
#include <hip/hip_bf16.h>

// ---------------------------------------------------------------------------
// Types for CDNA5 WMMA (wave32): 16x16x32 bf16 -> f32
// ---------------------------------------------------------------------------
typedef __attribute__((ext_vector_type(16))) __bf16    v16bf;
typedef __attribute__((ext_vector_type(8)))  float     v8f;
typedef __attribute__((ext_vector_type(8)))  unsigned  v8u;

// Problem constants (from reference setup_inputs)
#define BB 64      // batch
#define TT 20      // seq len
#define FF 2048    // feature dim
#define EE 512     // embed dim
#define GG 1024    // GRU hidden
#define VV 32000   // vocab

// ---------------------------------------------------------------------------
// A-tile loader: 16x32 bf16 (MxK), row-major source with leading dim lda.
// Contiguous k order (not the ISA table order; load_b_tile uses the same):
//   m = lane&15; lanes 0-15 hold K=0..15, lanes 16-31 K=16..31;
//   VGPR v in 0..7 holds K pair k0,k0+1 with k0 = hi*16 + 2*v
// ---------------------------------------------------------------------------
__device__ __forceinline__ v16bf load_a_tile(const __hip_bfloat16* __restrict__ A,
                                             int lda, int mBase, int kBase, int lane) {
  int m  = lane & 15;
  int hi = (lane >> 4) & 1;
  const unsigned* p = (const unsigned*)(A + (size_t)(mBase + m) * lda + kBase);
  v8u u;
#pragma unroll
  for (int v = 0; v < 8; ++v) {
    int k0 = hi * 16 + v * 2;   // hi*16+2v
    u[v] = p[k0 >> 1];
  }
  return __builtin_bit_cast(v16bf, u);
}

// ---------------------------------------------------------------------------
// B-tile loader: B = W^T where W is [N,K] row-major bf16; B is 32x16 (KxN).
// Contiguous k order (same as load_a_tile): column n = lane&15; lanes 0-15 hold K=0..15, lanes 16-31 K=16..31;
// VGPR v holds K pair (2v, 2v+1) within the lane-half. 32 contiguous bytes/lane.
// ---------------------------------------------------------------------------
__device__ __forceinline__ v16bf load_b_tile(const __hip_bfloat16* __restrict__ W,
                                             int ldw, int nBase, int kBase, int lane) {
  int n   = lane & 15;
  int hiK = ((lane >> 4) & 1) << 4;   // 0 or 16
  const unsigned* p = (const unsigned*)(W + (size_t)(nBase + n) * ldw + kBase + hiK);
  v8u u;
#pragma unroll
  for (int v = 0; v < 8; ++v) u[v] = p[v];
  return __builtin_bit_cast(v16bf, u);
}

// ---------------------------------------------------------------------------
// Per-row weight-norm scale: s[r] = g[r] / ||v[r,:]||_2 . One wave per row.
// ---------------------------------------------------------------------------
__global__ void row_scale_kernel(const float* __restrict__ v, const float* __restrict__ g,
                                 float* __restrict__ s, int R, int C) {
  int row  = (blockIdx.x * blockDim.x + threadIdx.x) >> 5;
  int lane = threadIdx.x & 31;
  if (row >= R) return;
  const float* p = v + (size_t)row * C;
  float acc = 0.f;
  for (int c = lane; c < C; c += 32) { float x = p[c]; acc += x * x; }
#pragma unroll
  for (int off = 16; off > 0; off >>= 1) acc += __shfl_down(acc, off);
  if (lane == 0) s[row] = g[row] * rsqrtf(acc);
}

// ---------------------------------------------------------------------------
// f32 -> bf16 conversion (grid-stride)
// ---------------------------------------------------------------------------
__global__ void f32_to_bf16_kernel(const float* __restrict__ in,
                                   __hip_bfloat16* __restrict__ out, long n) {
  long stride = (long)gridDim.x * blockDim.x;
  for (long i = (long)blockIdx.x * blockDim.x + threadIdx.x; i < n; i += stride)
    out[i] = __float2bfloat16(in[i]);
}

// ---------------------------------------------------------------------------
// x0[b,e] = bf16( s_img[e] * dot(features[b,:], img_v[e,:]) + img_b[e] )
// ---------------------------------------------------------------------------
__global__ void img_embed_kernel(const float* __restrict__ feat,
                                 const float* __restrict__ img_v,
                                 const float* __restrict__ s_img,
                                 const float* __restrict__ img_b,
                                 __hip_bfloat16* __restrict__ x0) {
  int idx = blockIdx.x * blockDim.x + threadIdx.x;
  if (idx >= BB * EE) return;
  int b = idx / EE, e = idx % EE;
  const float* f = feat + (size_t)b * FF;
  const float* w = img_v + (size_t)e * FF;
  float acc = 0.f;
  for (int k = 0; k < FF; ++k) acc += f[k] * w[k];
  x0[idx] = __float2bfloat16(acc * s_img[e] + img_b[e]);
}

// ---------------------------------------------------------------------------
// x[t,b,:] = bf16(emb_table[input_ids[b,t], :])  for t = 1..T-1
// ---------------------------------------------------------------------------
__global__ void embed_gather_kernel(const int* __restrict__ ids,
                                    const float* __restrict__ table,
                                    __hip_bfloat16* __restrict__ x) {
  long i = (long)blockIdx.x * blockDim.x + threadIdx.x;
  long total = (long)(TT - 1) * BB * EE;
  if (i >= total) return;
  int e = (int)(i % EE);
  long r = i / EE;
  int b = (int)(r % BB);
  int t = 1 + (int)(r / BB);
  int tok = ids[b * TT + t];
  x[((long)t * BB + b) * EE + e] = __float2bfloat16(table[(long)tok * EE + e]);
}

// ---------------------------------------------------------------------------
// Zero-init hidden state (f32 + bf16 copies); ws is poisoned 0xAA by harness.
// ---------------------------------------------------------------------------
__global__ void init_h_kernel(float* __restrict__ h, __hip_bfloat16* __restrict__ h_bf) {
  int i = blockIdx.x * blockDim.x + threadIdx.x;
  if (i >= BB * GG) return;
  h[i] = 0.f;
  h_bf[i] = __float2bfloat16(0.f);
}

// ---------------------------------------------------------------------------
// Generic WMMA GEMM:  C[M,N] = A[M,K] (bf16) @ W[N,K]^T (bf16) + bias[N]
// One 16x16 output tile per wave. M,N multiples of 16; K multiple of 32.
// ---------------------------------------------------------------------------
__global__ void wmma_gemm_bias_kernel(const __hip_bfloat16* __restrict__ A,
                                      const __hip_bfloat16* __restrict__ W,
                                      const float* __restrict__ bias,
                                      float* __restrict__ C,
                                      int M, int N, int K) {
  int wave = (blockIdx.x * blockDim.x + threadIdx.x) >> 5;
  int lane = threadIdx.x & 31;
  int nTiles = N >> 4;
  int mTile = wave / nTiles;
  int nTile = wave % nTiles;
  if (mTile * 16 >= M) return;                 // wave-uniform exit: EXEC stays full
  v8f acc = {};
  for (int k = 0; k < K; k += 32) {
    v16bf a = load_a_tile(A, K, mTile * 16, k, lane);
    v16bf b = load_b_tile(W, K, nTile * 16, k, lane);
    acc = __builtin_amdgcn_wmma_f32_16x16x32_bf16(false, a, false, b, (short)0, acc,
                                                  false, false);
  }
  int n    = nTile * 16 + (lane & 15);
  int mRow = mTile * 16 + ((lane >> 4) << 3);  // C layout: lanes16-31 -> rows +8
  float bv = bias[n];
#pragma unroll
  for (int v = 0; v < 8; ++v)
    C[(size_t)(mRow + v) * N + n] = acc[v] + bv;
}

// ---------------------------------------------------------------------------
// GRU gate fusion + state update (PyTorch GRUCell semantics):
//   r = sig(gi_r+gh_r); z = sig(gi_z+gh_z); n = tanh(gi_n + r*gh_n)
//   h' = (1-z)*n + z*h
// Writes h (f32, recurrence), h_bf (next-step GEMM A), hs_t (fc GEMM A row).
// ---------------------------------------------------------------------------
__global__ void gru_update_kernel(const float* __restrict__ gi,
                                  const float* __restrict__ gh,
                                  float* __restrict__ h,
                                  __hip_bfloat16* __restrict__ h_bf,
                                  __hip_bfloat16* __restrict__ hs_t) {
  int i = blockIdx.x * blockDim.x + threadIdx.x;
  if (i >= BB * GG) return;
  int b = i / GG, j = i % GG;
  const float* gib = gi + (size_t)b * 3 * GG;
  const float* ghb = gh + (size_t)b * 3 * GG;
  float r  = 1.f / (1.f + __expf(-(gib[j] + ghb[j])));
  float z  = 1.f / (1.f + __expf(-(gib[GG + j] + ghb[GG + j])));
  float nn = tanhf(gib[2 * GG + j] + r * ghb[2 * GG + j]);
  float hn = (1.f - z) * nn + z * h[i];
  h[i] = hn;
  __hip_bfloat16 hb = __float2bfloat16(hn);
  h_bf[i] = hb;
  hs_t[i] = hb;
}

// ---------------------------------------------------------------------------
// Vocab projection (dominant GEMM, 84 GFLOP):
//   out[b,t,n] = s_fc[n] * (hs[t*B+b,:] . fc_v[n,:]) + fc_b[n]
// Each wave: one 16-row M tile x 4 consecutive 16-col N tiles (A reused 4x).
// ---------------------------------------------------------------------------
__global__ void fc_wmma_kernel(const __hip_bfloat16* __restrict__ hs,   // [T*B, K]
                               const __hip_bfloat16* __restrict__ Wv,   // [V, K] bf16
                               const float* __restrict__ s,
                               const float* __restrict__ bias,
                               float* __restrict__ out) {               // [B, T, V]
  const int K = GG;
  const int NTPW = 4;                                    // N tiles per wave
  int wave = (blockIdx.x * blockDim.x + threadIdx.x) >> 5;
  int lane = threadIdx.x & 31;
  int nGroups = VV / (16 * NTPW);                        // 500
  int mTile = wave / nGroups;
  int ng    = wave % nGroups;
  if (mTile * 16 >= TT * BB) return;                     // wave-uniform
  int nBase = ng * 16 * NTPW;

  v8f acc[NTPW];
#pragma unroll
  for (int q = 0; q < NTPW; ++q) acc[q] = (v8f){};

  for (int k = 0; k < K; k += 32) {
    v16bf a = load_a_tile(hs, K, mTile * 16, k, lane);
#pragma unroll
    for (int q = 0; q < NTPW; ++q) {
      v16bf b = load_b_tile(Wv, K, nBase + 16 * q, k, lane);
      acc[q] = __builtin_amdgcn_wmma_f32_16x16x32_bf16(false, a, false, b, (short)0,
                                                       acc[q], false, false);
    }
  }

  int nl   = lane & 15;
  int mRow = mTile * 16 + ((lane >> 4) << 3);
#pragma unroll
  for (int q = 0; q < NTPW; ++q) {
    int n = nBase + 16 * q + nl;
    float sc = s[n], bv = bias[n];
#pragma unroll
    for (int v = 0; v < 8; ++v) {
      int m = mRow + v;                 // m = t*B + b
      int t = m >> 6;                   // /64
      int b = m & 63;
      out[((size_t)b * TT + t) * (size_t)VV + n] = acc[q][v] * sc + bv;
    }
  }
}

// ---------------------------------------------------------------------------
// Host-side orchestration
// ---------------------------------------------------------------------------
static inline size_t align256(size_t x) { return (x + 255) & ~(size_t)255; }

extern "C" void kernel_launch(void* const* d_in, const int* in_sizes, int n_in,
                              void* d_out, int out_size, void* d_ws, size_t ws_size,
                              hipStream_t stream) {
  (void)in_sizes; (void)n_in; (void)out_size; (void)ws_size;

  const float* features  = (const float*)d_in[0];
  const int*   input_ids = (const int*)  d_in[1];
  const float* emb_table = (const float*)d_in[2];
  const float* img_v     = (const float*)d_in[3];
  const float* img_g     = (const float*)d_in[4];
  const float* img_b     = (const float*)d_in[5];
  const float* W_ih      = (const float*)d_in[6];
  const float* W_hh      = (const float*)d_in[7];
  const float* b_ih      = (const float*)d_in[8];
  const float* b_hh      = (const float*)d_in[9];
  const float* fc_v      = (const float*)d_in[10];
  const float* fc_g      = (const float*)d_in[11];
  const float* fc_b      = (const float*)d_in[12];
  float* out = (float*)d_out;

  // Workspace layout
  char* ws = (char*)d_ws;
  size_t off = 0;
  __hip_bfloat16* fcv_bf = (__hip_bfloat16*)(ws + off); off += align256((size_t)VV * GG * 2);
  __hip_bfloat16* wih_bf = (__hip_bfloat16*)(ws + off); off += align256((size_t)3 * GG * EE * 2);
  __hip_bfloat16* whh_bf = (__hip_bfloat16*)(ws + off); off += align256((size_t)3 * GG * GG * 2);
  __hip_bfloat16* x_bf   = (__hip_bfloat16*)(ws + off); off += align256((size_t)TT * BB * EE * 2);
  __hip_bfloat16* hs_bf  = (__hip_bfloat16*)(ws + off); off += align256((size_t)TT * BB * GG * 2);
  float*          h_f32  = (float*)(ws + off);          off += align256((size_t)BB * GG * 4);
  __hip_bfloat16* h_bf   = (__hip_bfloat16*)(ws + off); off += align256((size_t)BB * GG * 2);
  float*          gi     = (float*)(ws + off);          off += align256((size_t)BB * 3 * GG * 4);
  float*          gh     = (float*)(ws + off);          off += align256((size_t)BB * 3 * GG * 4);
  float*          s_fc   = (float*)(ws + off);          off += align256((size_t)VV * 4);
  float*          s_img  = (float*)(ws + off);          off += align256((size_t)EE * 4);

  const int BLK = 256;

  // 1) weight-norm scales
  row_scale_kernel<<<VV / 8, BLK, 0, stream>>>(fc_v, fc_g, s_fc, VV, GG);
  row_scale_kernel<<<EE / 8, BLK, 0, stream>>>(img_v, img_g, s_img, EE, FF);

  // 2) f32 -> bf16 weight conversions
  f32_to_bf16_kernel<<<4096, BLK, 0, stream>>>(fc_v, fcv_bf, (long)VV * GG);
  f32_to_bf16_kernel<<<2048, BLK, 0, stream>>>(W_ih, wih_bf, (long)3 * GG * EE);
  f32_to_bf16_kernel<<<2048, BLK, 0, stream>>>(W_hh, whh_bf, (long)3 * GG * GG);

  // 3) sequence inputs: step 0 = image embedding, steps 1..T-1 = token embeddings
  img_embed_kernel<<<(BB * EE) / BLK, BLK, 0, stream>>>(features, img_v, s_img, img_b, x_bf);
  {
    long total = (long)(TT - 1) * BB * EE;
    embed_gather_kernel<<<(int)((total + BLK - 1) / BLK), BLK, 0, stream>>>(
        input_ids, emb_table, x_bf);
  }

  // 4) GRU recurrence
  init_h_kernel<<<(BB * GG) / BLK, BLK, 0, stream>>>(h_f32, h_bf);
  const int gateTiles  = (BB / 16) * ((3 * GG) / 16);   // 4 * 192 = 768 waves
  const int gateBlocks = gateTiles / (BLK / 32);        // 96
  for (int t = 0; t < TT; ++t) {
    wmma_gemm_bias_kernel<<<gateBlocks, BLK, 0, stream>>>(
        x_bf + (size_t)t * BB * EE, wih_bf, b_ih, gi, BB, 3 * GG, EE);
    wmma_gemm_bias_kernel<<<gateBlocks, BLK, 0, stream>>>(
        h_bf, whh_bf, b_hh, gh, BB, 3 * GG, GG);
    gru_update_kernel<<<(BB * GG) / BLK, BLK, 0, stream>>>(
        gi, gh, h_f32, h_bf, hs_bf + (size_t)t * BB * GG);
  }

  // 5) vocab projection (dominant WMMA GEMM) with weight-norm scale + bias,
  //    scattered directly into [B,T,V] output order.
  const int fcWaves  = ((TT * BB) / 16) * (VV / 64);    // 80 * 500 = 40000
  const int fcBlocks = fcWaves / (BLK / 32);            // 5000
  fc_wmma_kernel<<<fcBlocks, BLK, 0, stream>>>(hs_bf, fcv_bf, s_fc, fc_b, out);
}
